// VectorQuantizer_15375982920157
// MI455X (gfx1250) — matrix operand order repaired and checked by reading, not compiled, not run on hardware
//
#include <hip/hip_runtime.h>

// VQ quantizer for MI455X (gfx1250, wave32, WMMA).
// d_in[0]: x  [64*512*256] f32   (N=32768 rows, D=256)
// d_in[1]: E  [256*1024]   f32   (embeddings, [D][K])
// d_out  : out [32768*256] f32, then loss scalar at offset 8388608
// d_ws   : >= ~2.1 MB scratch (eT f32 1MB, eT hi/lo bf16 0.5MB each,
//          enorm 4KB, per-block loss partials 4KB)

typedef __attribute__((ext_vector_type(16))) __bf16 v16bf;
typedef __attribute__((ext_vector_type(8)))  __bf16 v8bf;
typedef __attribute__((ext_vector_type(8)))  float  v8f;

#define DDIM     256
#define KCODES   1024
#define BM       32
#define NTHREADS 256

// ---------------------------------------------------------------------------
// Prep: transpose E [D][K] -> eT [K][D] (f32 + bf16 hi/lo split), and |e_k|^2
// ---------------------------------------------------------------------------
__global__ void __launch_bounds__(NTHREADS)
vq_prep(const float* __restrict__ E,
        float* __restrict__ eTf,
        __bf16* __restrict__ eThi,
        __bf16* __restrict__ eTlo,
        float* __restrict__ enorm)
{
    __shared__ float red[NTHREADS];
    const int k = blockIdx.x;    // code 0..1023
    const int d = threadIdx.x;   // dim  0..255

    float v = E[(size_t)d * KCODES + k];
    eTf[(size_t)k * DDIM + d] = v;
    __bf16 h = (__bf16)v;
    eThi[(size_t)k * DDIM + d] = h;
    eTlo[(size_t)k * DDIM + d] = (__bf16)(v - (float)h);

    red[d] = v * v;
    __syncthreads();
    for (int s = NTHREADS / 2; s > 0; s >>= 1) {
        if (d < s) red[d] += red[d + s];
        __syncthreads();
    }
    if (d == 0) enorm[k] = red[0];
}

// ---------------------------------------------------------------------------
// Main: per 32-row block, compute dists to all 1024 codes via split-bf16 WMMA,
// argmin per row, gather f32 codeword to out, accumulate squared error.
// ---------------------------------------------------------------------------
__global__ void __launch_bounds__(NTHREADS)
vq_main(const float* __restrict__ x,
        const float* __restrict__ eTf,
        const __bf16* __restrict__ eThi,
        const __bf16* __restrict__ eTlo,
        const float* __restrict__ enorm,
        float* __restrict__ out,
        float* __restrict__ partials)
{
    __shared__ __bf16 sAhi[BM * DDIM];     // 16 KB
    __shared__ __bf16 sAlo[BM * DDIM];     // 16 KB
    __shared__ float  sXnp[BM][8];         // per-row norm partials
    __shared__ float  sXn[BM];             // per-row |x|^2
    __shared__ float  sRd[BM][64];         // 8 KB  argmin slots (dist)
    __shared__ int    sRi[BM][64];         // 8 KB  argmin slots (idx)
    __shared__ int    sBest[BM];
    __shared__ float  sLoss[NTHREADS];     // 1 KB

    const int    tid     = threadIdx.x;
    const size_t rowBase = (size_t)blockIdx.x * BM;

    // ---- stage x rows into LDS as bf16 hi/lo; per-row norm partials ----
    {
        const int row  = tid >> 3;          // 0..31
        const int dseg = (tid & 7) * 32;    // 8 threads per row
        const float* xr = x + (rowBase + row) * DDIM + dseg;
        float acc = 0.f;
        #pragma unroll
        for (int k = 0; k < 32; k++) {
            float v = xr[k];
            __bf16 h = (__bf16)v;
            sAhi[row * DDIM + dseg + k] = h;
            sAlo[row * DDIM + dseg + k] = (__bf16)(v - (float)h);
            acc += v * v;
        }
        sXnp[row][tid & 7] = acc;
    }
    __syncthreads();
    if (tid < BM) {
        float s = 0.f;
        #pragma unroll
        for (int k = 0; k < 8; k++) s += sXnp[tid][k];
        sXn[tid] = s;
    }
    __syncthreads();

    // ---- wave / lane geometry ----
    const int lane  = tid & 31;
    const int wave  = tid >> 5;       // 0..7
    const int rtile = wave & 1;       // row tile (16 rows each)
    const int ctile = wave >> 1;      // col tile 0..3 within 64-code chunk
    const int ln    = lane & 15;      // column within 16
    const int half  = lane >> 4;      // 0/1: which K-half this lane holds
    const int abase = half * 8;       // accumulator row base: D row = 8*half + j
    const int koff  = half * 16;      // A/B-frag K base offset (contiguous k = 16*half + s)

    // ---- hoist A fragments (hi/lo) for all 8 K-chunks into registers ----
    v16bf Ahi[8], Alo[8];
    const int arow = rtile * 16 + ln;  // lanes 0-15 and 16-31 both map M=ln
    #pragma unroll
    for (int dc = 0; dc < 8; dc++) {
        const int d0 = dc * 32;
        v8bf p0 = *(const v8bf*)&sAhi[arow * DDIM + d0 + koff];
        v8bf p1 = *(const v8bf*)&sAhi[arow * DDIM + d0 + koff + 8];
        v8bf q0 = *(const v8bf*)&sAlo[arow * DDIM + d0 + koff];
        v8bf q1 = *(const v8bf*)&sAlo[arow * DDIM + d0 + koff + 8];
        #pragma unroll
        for (int i = 0; i < 8; i++) {
            Ahi[dc][i]     = p0[i];
            Ahi[dc][8 + i] = p1[i];
            Alo[dc][i]     = q0[i];
            Alo[dc][8 + i] = q1[i];
        }
    }

    float xn[8];
    #pragma unroll
    for (int j = 0; j < 8; j++) xn[j] = sXn[rtile * 16 + abase + j];

    float bestd[8];
    int   besti[8];
    #pragma unroll
    for (int j = 0; j < 8; j++) { bestd[j] = 3.4e38f; besti[j] = 0; }

    // ---- sweep all codes in 64-wide chunks; 16x16 tile per wave per chunk ----
    for (int n0 = 0; n0 < KCODES; n0 += 64) {
        const int code = n0 + ctile * 16 + ln;
        const __bf16* bh = eThi + (size_t)code * DDIM + koff;
        const __bf16* bl = eTlo + (size_t)code * DDIM + koff;
        v8f acc = {};
        #pragma unroll
        for (int dc = 0; dc < 8; dc++) {
            v16bf Bhi = *(const v16bf*)(bh + dc * 32);
            v16bf Blo = *(const v16bf*)(bl + dc * 32);
            // split-bf16: x.e ~= lo*hi + hi*lo + hi*hi (f32 accumulate)
            acc = __builtin_amdgcn_wmma_f32_16x16x32_bf16(
                false, Alo[dc], false, Bhi, (short)0, acc, false, false);
            acc = __builtin_amdgcn_wmma_f32_16x16x32_bf16(
                false, Ahi[dc], false, Blo, (short)0, acc, false, false);
            acc = __builtin_amdgcn_wmma_f32_16x16x32_bf16(
                false, Ahi[dc], false, Bhi, (short)0, acc, false, false);
        }
        const float en = enorm[code];
        #pragma unroll
        for (int j = 0; j < 8; j++) {
            float dist = xn[j] + en - 2.0f * acc[j];
            if (dist < bestd[j]) { bestd[j] = dist; besti[j] = code; }
        }
    }

    // ---- per-lane mins -> LDS slots ----
    #pragma unroll
    for (int j = 0; j < 8; j++) {
        const int row  = rtile * 16 + abase + j;
        const int slot = ctile * 16 + ln;
        sRd[row][slot] = bestd[j];
        sRi[row][slot] = besti[j];
    }
    __syncthreads();

    // ---- per-row argmin over 64 slots ----
    if (tid < BM) {
        float bd = sRd[tid][0];
        int   bi = sRi[tid][0];
        for (int s = 1; s < 64; s++) {
            float d2 = sRd[tid][s];
            if (d2 < bd) { bd = d2; bi = sRi[tid][s]; }
        }
        sBest[tid] = bi;
    }
    __syncthreads();

    // ---- gather f32 codewords to out; accumulate (q - x)^2 ----
    float lsum = 0.f;
    for (int e = tid; e < BM * DDIM; e += NTHREADS) {
        const int row = e >> 8;     // e = i*256 + tid -> row=i, d=tid (coalesced)
        const int d   = e & 255;
        const float q  = eTf[(size_t)sBest[row] * DDIM + d];
        const float xv = x[(rowBase + row) * DDIM + d];
        out[(rowBase + row) * DDIM + d] = q;
        const float df = q - xv;
        lsum += df * df;
    }
    sLoss[tid] = lsum;
    __syncthreads();
    for (int s = NTHREADS / 2; s > 0; s >>= 1) {
        if (tid < s) sLoss[tid] += sLoss[tid + s];
        __syncthreads();
    }
    if (tid == 0) partials[blockIdx.x] = sLoss[0];
}

// ---------------------------------------------------------------------------
// Loss: deterministic fixed-order reduction of 1024 block partials.
// loss = 1.25 * total / (B*D) = 1.25 * total / 16384
// ---------------------------------------------------------------------------
__global__ void __launch_bounds__(NTHREADS)
vq_loss(const float* __restrict__ partials, float* __restrict__ lossOut)
{
    __shared__ float red[NTHREADS];
    const int t = threadIdx.x;
    float s = partials[t] + partials[t + 256] + partials[t + 512] + partials[t + 768];
    red[t] = s;
    __syncthreads();
    for (int k = NTHREADS / 2; k > 0; k >>= 1) {
        if (t < k) red[t] += red[t + k];
        __syncthreads();
    }
    if (t == 0) lossOut[0] = red[0] * (1.25f / 16384.0f);
}

// ---------------------------------------------------------------------------
extern "C" void kernel_launch(void* const* d_in, const int* in_sizes, int n_in,
                              void* d_out, int out_size, void* d_ws, size_t ws_size,
                              hipStream_t stream)
{
    (void)in_sizes; (void)n_in; (void)out_size; (void)ws_size;

    const float* x = (const float*)d_in[0];   // 64*512*256 f32
    const float* E = (const float*)d_in[1];   // 256*1024   f32
    float* out = (float*)d_out;               // 8388608 f32 + 1 loss

    char* ws = (char*)d_ws;
    float*  eTf      = (float*)(ws);                                  // 1 MB
    __bf16* eThi     = (__bf16*)(ws + (1u << 20));                    // 512 KB
    __bf16* eTlo     = (__bf16*)(ws + (1u << 20) + (1u << 19));       // 512 KB
    float*  enorm    = (float*)(ws + (1u << 21));                     // 4 KB
    float*  partials = enorm + KCODES;                                // 4 KB

    const int NROWS = 64 * 512;   // 32768

    vq_prep<<<KCODES, NTHREADS, 0, stream>>>(E, eTf, eThi, eTlo, enorm);
    vq_main<<<NROWS / BM, NTHREADS, 0, stream>>>(x, eTf, eThi, eTlo, enorm,
                                                 out, partials);
    vq_loss<<<1, NTHREADS, 0, stream>>>(partials, out + (size_t)NROWS * DDIM);
}
